// CAFormer_63677185130571
// MI455X (gfx1250) — compile-verified, matrix operand order checked by reading, not run on hardware
//
#include <hip/hip_runtime.h>
#include <hip/hip_bf16.h>
#include <math.h>

// Problem constants
#define Bn   4
#define DIM  64
#define NH   8
#define HD   8
#define Hp   256
#define Wp   256
#define HW   65536   // 256*256

typedef _Float16 v16h __attribute__((ext_vector_type(16)));
typedef float    v8f  __attribute__((ext_vector_type(8)));
typedef unsigned int u32x4 __attribute__((ext_vector_type(4)));
typedef unsigned int u32x8 __attribute__((ext_vector_type(8)));

// LDS strides chosen so 16 consecutive "n/m" lanes hit distinct banks.
#define LDK 260   // halves; 520B row stride -> bank = 2n mod 64, conflict free
#define LDW 72    // halves; small-tile stride

// Workspace layout (bytes)
//  qkv f16 : B*192*HW*2            = 100,663,296
//  attn f32: B*64*HW*4             =  67,108,864
//  mix  f32: B*64*HW*4             =  67,108,864
//  stats f32: gx[256] ss[512] invn[512] cwaw[8]
#define OFF_QKV   ((size_t)0)
#define OFF_ATTN  ((size_t)100663296)
#define OFF_MIX   ((size_t)167772160)
#define OFF_STATS ((size_t)234881024)
// float offsets inside stats
#define SGX   0
#define SSS   256
#define SINV  768
#define SCW   1280

static __device__ inline v8f wmma16(v16h a, v16h b, v8f c) {
  return __builtin_amdgcn_wmma_f32_16x16x32_f16(false, a, false, b, (short)0, c,
                                                false, false);
}

// Load a 16x32 f16 A-fragment (or B^T-fragment) from an LDS/global row-major
// tile: element [m][k], row stride ld (halves). Layout per ISA 7.12.2:
// lane L: m=L%16, half=L/16; VGPR j<4: k=k0+8*half+2j; j>=4: k=k0+16+8*half+2(j-4)
static __device__ inline v16h fragA(const _Float16* base, int row0, int ld,
                                    int k0, int lane) {
  int m = lane & 15, h = lane >> 4;
  const _Float16* r = base + (size_t)(row0 + m) * ld;
  v16h a;
#pragma unroll
  for (int j = 0; j < 4; ++j) {
    int k = k0 + 8 * h + 2 * j;
    a[2 * j] = r[k];
    a[2 * j + 1] = r[k + 1];
  }
#pragma unroll
  for (int j = 0; j < 4; ++j) {
    int k = k0 + 16 + 8 * h + 2 * j;
    a[8 + 2 * j] = r[k];
    a[8 + 2 * j + 1] = r[k + 1];
  }
  return a;
}

// Q fragment straight from a global 256x256 f16 plane; dir=0 row-major rows,
// dir=1 reads the transposed matrix (columns of the plane).
static __device__ inline v16h fragQ(const _Float16* qp, int row0, int k0,
                                    int dir, int lane) {
  int m = lane & 15, h = lane >> 4;
  v16h a;
  if (dir == 0) {
    const _Float16* r = qp + (size_t)(row0 + m) * Wp;
#pragma unroll
    for (int j = 0; j < 4; ++j) {
      int k = k0 + 8 * h + 2 * j;
      a[2 * j] = r[k];
      a[2 * j + 1] = r[k + 1];
    }
#pragma unroll
    for (int j = 0; j < 4; ++j) {
      int k = k0 + 16 + 8 * h + 2 * j;
      a[8 + 2 * j] = r[k];
      a[8 + 2 * j + 1] = r[k + 1];
    }
  } else {
    int col = row0 + m;
#pragma unroll
    for (int j = 0; j < 4; ++j) {
      int k = k0 + 8 * h + 2 * j;
      a[2 * j] = qp[(size_t)k * Wp + col];
      a[2 * j + 1] = qp[(size_t)(k + 1) * Wp + col];
    }
#pragma unroll
    for (int j = 0; j < 4; ++j) {
      int k = k0 + 16 + 8 * h + 2 * j;
      a[8 + 2 * j] = qp[(size_t)k * Wp + col];
      a[8 + 2 * j + 1] = qp[(size_t)(k + 1) * Wp + col];
    }
  }
  return a;
}

// ---------------------------------------------------------------------------
// TDM: DMA a contiguous 256x256 f16 plane (row-major) from global into LDS,
// with HW row padding: pad_interval=6 (pad after each 128 DWORDs = one 512B
// row), pad_amount=1 (2 DWORDs = 4 halves) -> LDS row stride = 260 halves,
// matching LDK. D# per CDNA5 ISA. Issued by one wave;
// tracked with TENSORcnt.
// ---------------------------------------------------------------------------
static __device__ inline void tdm_load_plane(unsigned lds_byte_off,
                                             const _Float16* gptr) {
  union { unsigned u[4]; u32x4 v; } g0;
  union { unsigned u[8]; u32x8 v; } g1;
  unsigned long long ga = (unsigned long long)(const void*)gptr;
  g0.u[0] = 1u;                        // count=1, is_restore=0, gather off
  g0.u[1] = lds_byte_off;              // lds_addr (bytes)
  g0.u[2] = (unsigned)ga;              // global_addr[31:0]
  g0.u[3] = (unsigned)((ga >> 32) & 0x1FFFFFFu) | (2u << 30);  // [56:32]|type=2
  g1.u[0] = (1u << 16)                 // data_size = 2 bytes
          | (1u << 20)                 // pad_enable
          | (6u << 22)                 // pad_interval: 128 DWORDs (one row)
          | (1u << 25);                // pad_amount: 2 DWORDs (4 halves)
  g1.u[1] = (256u & 0xffffu) << 16;    // tensor_dim0[15:0] = 256
  g1.u[2] = (256u & 0xffffu) << 16;    // tensor_dim0[31:16]=0 | tensor_dim1 lo
  g1.u[3] = (256u << 16);              // tensor_dim1 hi=0 | tile_dim0 = 256
  g1.u[4] = 256u;                      // tile_dim1 = 256, tile_dim2 = 0
  g1.u[5] = 256u;                      // tensor_dim0_stride[31:0] = 256
  g1.u[6] = 0u;                        // stride0 hi | tensor_dim1_stride lo
  g1.u[7] = 0u;
  asm volatile("tensor_load_to_lds %0, %1" :: "s"(g0.v), "s"(g1.v) : "memory");
}

// ---------------------------------------------------------------------------
// Kernel 1: qkv = wqkv @ x (per pixel), store f16; fused channel-sum (for the
// gating mean) and q/k channel sum-of-squares (for the fused l2norm scalars).
// grid (1024, B), 128 threads (4 waves). Each block: 64 pixels x 192 outputs.
// ---------------------------------------------------------------------------
__global__ void k_qkv(const float* __restrict__ x,
                      const float* __restrict__ wqkv,
                      _Float16* __restrict__ qkv16,
                      float* __restrict__ stats) {
  __shared__ _Float16 xs[64 * LDW];    // x tile, [pixel][channel]
  __shared__ _Float16 wsm[192 * LDW];  // wqkv f16, [o][c]
  __shared__ float sss[128];           // q/k sum-of-squares partials
  int t = threadIdx.x, b = blockIdx.y;
  int P0 = blockIdx.x * 64;
  int lane = t & 31, wv = t >> 5;

  for (int i = t; i < 192 * 64; i += 128) {
    int o = i >> 6, c = i & 63;
    wsm[o * LDW + c] = (_Float16)wqkv[i];
  }
  if (t < 128) sss[t] = 0.f;
  for (int i = t; i < 64 * 64; i += 128) {
    int c = i >> 6, p = i & 63;
    xs[p * LDW + c] = (_Float16)x[((size_t)(b * DIM + c) << 16) + P0 + p];
  }
  __syncthreads();

  // channel partial sums for the gating mean
  if (t < 64) {
    float s = 0.f;
#pragma unroll
    for (int p = 0; p < 64; ++p) s += (float)xs[p * LDW + t];
    atomicAdd(&stats[SGX + b * DIM + t], s);
  }

  int n = lane & 15, h = lane >> 4;
  v16h b0 = fragA(xs, wv * 16, LDW, 0, lane);
  v16h b1 = fragA(xs, wv * 16, LDW, 32, lane);
#pragma unroll
  for (int mt = 0; mt < 12; ++mt) {
    v8f acc = {0.f, 0.f, 0.f, 0.f, 0.f, 0.f, 0.f, 0.f};
    v16h a0 = fragA(wsm, mt * 16, LDW, 0, lane);
    v16h a1 = fragA(wsm, mt * 16, LDW, 32, lane);
    acc = wmma16(a0, b0, acc);
    acc = wmma16(a1, b1, acc);
    int pix = P0 + wv * 16 + n;
#pragma unroll
    for (int r = 0; r < 8; ++r) {
      int ch = mt * 16 + 8 * h + r;
      float v = acc[r];
      qkv16[((size_t)(b * 192 + ch) << 16) + pix] = (_Float16)v;
      if (ch < 128) atomicAdd(&sss[ch], v * v);  // ds_add_f32
    }
  }
  __syncthreads();
  if (t < 128) atomicAdd(&stats[SSS + b * 128 + t], sss[t]);
}

// ---------------------------------------------------------------------------
// Kernel 2: tiny gating head + inverse norms. 1 block, 256 threads.
// ---------------------------------------------------------------------------
__global__ void k_gate(const float* __restrict__ g1_w,
                       const float* __restrict__ g1_b,
                       const float* __restrict__ g2_w,
                       const float* __restrict__ g2_b,
                       float* __restrict__ stats) {
  __shared__ float sh1[Bn][16];
  __shared__ float slog[Bn][2];
  int t = threadIdx.x;
  for (int i = t; i < Bn * 128; i += 256) {
    float nrm = sqrtf(stats[SSS + i]);
    stats[SINV + i] = 1.f / fmaxf(nrm, 1e-12f);
  }
  if (t < Bn * 16) {
    int b = t >> 4, j = t & 15;
    float acc = g1_b[j];
#pragma unroll
    for (int c = 0; c < DIM; ++c)
      acc += g1_w[j * DIM + c] * (stats[SGX + b * DIM + c] * (1.f / 65536.f));
    sh1[b][j] = fmaxf(acc, 0.f);
  }
  __syncthreads();
  if (t < Bn * 2) {
    int b = t >> 1, o = t & 1;
    float acc = g2_b[o];
#pragma unroll
    for (int j = 0; j < 16; ++j) acc += g2_w[o * 16 + j] * sh1[b][j];
    slog[b][o] = acc;
  }
  __syncthreads();
  if (t < Bn) {
    float l0 = slog[t][0], l1 = slog[t][1];
    float m = fmaxf(l0, l1);
    float e0 = __expf(l0 - m), e1 = __expf(l1 - m);
    float inv = 1.f / (e0 + e1);
    stats[SCW + 2 * t + 0] = e0 * inv;  // conv weight
    stats[SCW + 2 * t + 1] = e1 * inv;  // attn weight
  }
}

// ---------------------------------------------------------------------------
// Kernel 3: per-slice 256x256 attention (dir=0 horizontal, dir=1 vertical ==
// same math on the transposed plane). One block per slice, 128 threads
// (4 waves), whole K and V^T resident in LDS (2*133KB + 33KB < 320KB/WGP).
// The contiguous-plane stage goes through the Tensor Data Mover.
// ---------------------------------------------------------------------------
__global__ void k_attn(const _Float16* __restrict__ qkv16,
                       const float* __restrict__ scale,
                       const float* __restrict__ stats,
                       float* __restrict__ attn, int dir) {
  extern __shared__ _Float16 sm[];
  _Float16* Kmat = sm;                // [z][y]   256 x LDK
  _Float16* VT = sm + 256 * LDK;      // [y][z]   256 x LDK
  _Float16* Pb = VT + 256 * LDK;      // per-wave prob tiles: 4 x 16 x LDK
  int t = threadIdx.x, lane = t & 31, wv = t >> 5;
  int s = blockIdx.x, b = s >> 6, ch = s & 63, nh = ch >> 3;
  const _Float16* qp = qkv16 + ((size_t)(b * 192 + ch) << 16);
  const _Float16* kp = qkv16 + ((size_t)(b * 192 + 64 + ch) << 16);
  const _Float16* vp = qkv16 + ((size_t)(b * 192 + 128 + ch) << 16);
  // l2norm of q,k folded into one per-slice logit scale
  float st = scale[nh] * stats[SINV + b * 128 + ch] *
             stats[SINV + b * 128 + 64 + ch];
  float* op = attn + ((size_t)(b * DIM + ch) << 16);

  __builtin_prefetch((const void*)qp, 0, 3);  // global_prefetch_b8

  if (dir == 0) {
    // K is consumed row-major: TDM DMA with HW padding to stride LDK.
    if (wv == 0) tdm_load_plane(0u, kp);
    for (int l = t; l < HW; l += 128) {  // V^T: gather transpose
      int y = l >> 8, z = l & 255;
      VT[y * LDK + z] = vp[(size_t)z * Wp + y];
    }
  } else {
    // V'^T == vplane row-major: TDM DMA.
    if (wv == 0) tdm_load_plane((unsigned)(256 * LDK * 2), vp);
    for (int l = t; l < HW; l += 128) {  // K' = kplane^T: gather transpose
      int z = l >> 8, xx = l & 255;
      Kmat[z * LDK + xx] = kp[(size_t)xx * Wp + z];
    }
  }
  if (wv == 0) __builtin_amdgcn_s_wait_tensorcnt(0);
  __syncthreads();

  _Float16* Pw = Pb + wv * 16 * LDK;
  int n = lane & 15, h = lane >> 4;

  for (int xt = wv; xt < 16; xt += 4) {
    int x0 = xt * 16;
    v16h aq[8];
#pragma unroll
    for (int kc = 0; kc < 8; ++kc) aq[kc] = fragQ(qp, x0, kc * 32, dir, lane);

    // logits: 16x256 tile held in registers (16 C-fragments)
    v8f lg[16];
#pragma unroll
    for (int zt = 0; zt < 16; ++zt) {
      v8f acc = {0.f, 0.f, 0.f, 0.f, 0.f, 0.f, 0.f, 0.f};
#pragma unroll
      for (int kc = 0; kc < 8; ++kc) {
        v16h bf = fragA(Kmat, zt * 16, LDK, kc * 32, lane);
        acc = wmma16(aq[kc], bf, acc);
      }
      lg[zt] = acc;
    }

    // softmax per row: row m=8*h+r lives in VGPR slot r across 16 lanes of
    // this lane-half (C layout) x 16 zt registers -> shuffle-reduce xor<16.
#pragma unroll
    for (int r = 0; r < 8; ++r) {
      float mx = -3.4e38f;
#pragma unroll
      for (int zt = 0; zt < 16; ++zt) mx = fmaxf(mx, lg[zt][r]);
#pragma unroll
      for (int msk = 1; msk < 16; msk <<= 1)
        mx = fmaxf(mx, __shfl_xor(mx, msk, 32));
      float sum = 0.f;
#pragma unroll
      for (int zt = 0; zt < 16; ++zt) {
        float p = __expf(st * (lg[zt][r] - mx));
        lg[zt][r] = p;
        sum += p;
      }
#pragma unroll
      for (int msk = 1; msk < 16; msk <<= 1) sum += __shfl_xor(sum, msk, 32);
      float inv = 1.f / sum;
      int mrow = 8 * h + r;
#pragma unroll
      for (int zt = 0; zt < 16; ++zt)
        Pw[mrow * LDK + zt * 16 + n] = (_Float16)(lg[zt][r] * inv);
    }
    // re-fragment probabilities as A (wave-private LDS; DS ops are in-order)
#pragma unroll
    for (int kc = 0; kc < 8; ++kc) aq[kc] = fragA(Pw, 0, LDK, kc * 32, lane);

    // out tile = P @ V
#pragma unroll
    for (int yt = 0; yt < 16; ++yt) {
      v8f acc = {0.f, 0.f, 0.f, 0.f, 0.f, 0.f, 0.f, 0.f};
#pragma unroll
      for (int kc = 0; kc < 8; ++kc) {
        v16h bf = fragA(VT, yt * 16, LDK, kc * 32, lane);
        acc = wmma16(aq[kc], bf, acc);
      }
      int y0 = yt * 16;
      if (dir == 0) {
#pragma unroll
        for (int r = 0; r < 8; ++r)
          op[(size_t)(x0 + 8 * h + r) * Wp + y0 + n] = acc[r];
      } else {  // computed on M^T -> scatter-add back transposed
#pragma unroll
        for (int r = 0; r < 8; ++r) {
          size_t a = (size_t)(y0 + n) * Wp + x0 + 8 * h + r;
          op[a] += acc[r];
        }
      }
    }
  }
}

// ---------------------------------------------------------------------------
// Kernel 4: depthwise 3x3 + 5x5 conv (one image row per block, halo in LDS)
// and the gated combine with the attention features.
// ---------------------------------------------------------------------------
__global__ void k_mix(const float* __restrict__ x, const float* __restrict__ w3,
                      const float* __restrict__ b3, const float* __restrict__ w5,
                      const float* __restrict__ b5,
                      const float* __restrict__ attn,
                      const float* __restrict__ stats,
                      float* __restrict__ mix) {
  __shared__ float rows[5][264];
  int t = threadIdx.x;
  int idx = blockIdx.x;
  int b = idx >> 14, c = (idx >> 8) & 63, r = idx & 255;
  const float* xp = x + ((size_t)(b * DIM + c) << 16);
#pragma unroll
  for (int rr = 0; rr < 5; ++rr) {
    int xr = r + rr - 2;
    for (int col = t; col < 260; col += 256) {
      int xc = col - 2;
      float v = 0.f;
      if (xr >= 0 && xr < Hp && xc >= 0 && xc < Wp) v = xp[xr * Wp + xc];
      rows[rr][col] = v;
    }
  }
  __syncthreads();
  float acc = b3[c] + b5[c];
#pragma unroll
  for (int i = 0; i < 3; ++i)
#pragma unroll
    for (int j = 0; j < 3; ++j)
      acc += w3[c * 9 + i * 3 + j] * rows[1 + i][t + j + 1];
#pragma unroll
  for (int i = 0; i < 5; ++i)
#pragma unroll
    for (int j = 0; j < 5; ++j)
      acc += w5[c * 25 + i * 5 + j] * rows[i][t + j];
  size_t pix = ((size_t)(b * DIM + c) << 16) + (size_t)r * Wp + t;
  float cw = stats[SCW + 2 * b], aw = stats[SCW + 2 * b + 1];
  mix[pix] = cw * acc + aw * attn[pix];
}

// ---------------------------------------------------------------------------
// Kernel 5: final 64x64 projection (WMMA) + bias.
// grid (1024, B), 128 threads; 64 pixels x 64 outputs per block.
// ---------------------------------------------------------------------------
__global__ void k_proj(const float* __restrict__ wp,
                       const float* __restrict__ bp,
                       const float* __restrict__ mix,
                       float* __restrict__ out) {
  __shared__ _Float16 wps[64 * LDW];  // [o][c]
  __shared__ _Float16 ft[64 * LDW];   // [pixel][c]
  int t = threadIdx.x, b = blockIdx.y;
  int P0 = blockIdx.x * 64;
  for (int i = t; i < 4096; i += 128) {
    int o = i >> 6, c = i & 63;
    wps[o * LDW + c] = (_Float16)wp[i];
  }
  for (int i = t; i < 4096; i += 128) {
    int c = i >> 6, p = i & 63;
    ft[p * LDW + c] = (_Float16)mix[((size_t)(b * DIM + c) << 16) + P0 + p];
  }
  __syncthreads();
  int lane = t & 31, wv = t >> 5, n = lane & 15, h = lane >> 4;
  v16h b0 = fragA(ft, wv * 16, LDW, 0, lane);
  v16h b1 = fragA(ft, wv * 16, LDW, 32, lane);
#pragma unroll
  for (int mt = 0; mt < 4; ++mt) {
    v8f acc = {0.f, 0.f, 0.f, 0.f, 0.f, 0.f, 0.f, 0.f};
    v16h a0 = fragA(wps, mt * 16, LDW, 0, lane);
    v16h a1 = fragA(wps, mt * 16, LDW, 32, lane);
    acc = wmma16(a0, b0, acc);
    acc = wmma16(a1, b1, acc);
    int pix = P0 + wv * 16 + n;
#pragma unroll
    for (int r = 0; r < 8; ++r) {
      int o = mt * 16 + 8 * h + r;
      out[((size_t)(b * DIM + o) << 16) + pix] = acc[r] + bp[o];
    }
  }
}

extern "C" void kernel_launch(void* const* d_in, const int* in_sizes, int n_in,
                              void* d_out, int out_size, void* d_ws,
                              size_t ws_size, hipStream_t stream) {
  const float* x = (const float*)d_in[0];
  const float* w3 = (const float*)d_in[1];
  const float* b3 = (const float*)d_in[2];
  const float* w5 = (const float*)d_in[3];
  const float* b5 = (const float*)d_in[4];
  const float* wqkv = (const float*)d_in[5];
  const float* scale = (const float*)d_in[6];
  const float* g1_w = (const float*)d_in[7];
  const float* g1_b = (const float*)d_in[8];
  const float* g2_w = (const float*)d_in[9];
  const float* g2_b = (const float*)d_in[10];
  const float* wp = (const float*)d_in[11];
  const float* bp = (const float*)d_in[12];
  float* out = (float*)d_out;

  char* ws = (char*)d_ws;
  _Float16* qkv16 = (_Float16*)(ws + OFF_QKV);
  float* attn = (float*)(ws + OFF_ATTN);
  float* mixb = (float*)(ws + OFF_MIX);
  float* stats = (float*)(ws + OFF_STATS);

  // zero gx + sumsq accumulators (768 floats)
  hipMemsetAsync(stats, 0, 768 * sizeof(float), stream);

  k_qkv<<<dim3(1024, Bn), 128, 0, stream>>>(x, wqkv, qkv16, stats);
  k_gate<<<1, 256, 0, stream>>>(g1_w, g1_b, g2_w, g2_b, stats);

  const size_t attn_smem = (size_t)(2 * 256 * LDK + 4 * 16 * LDK) * sizeof(_Float16);
  k_attn<<<Bn * DIM, 128, attn_smem, stream>>>(qkv16, scale, stats, attn, 0);
  k_attn<<<Bn * DIM, 128, attn_smem, stream>>>(qkv16, scale, stats, attn, 1);

  k_mix<<<Bn * DIM * Hp, 256, 0, stream>>>(x, w3, b3, w5, b5, attn, stats, mixb);
  k_proj<<<dim3(1024, Bn), 128, 0, stream>>>(wp, bp, mixb, out);
}
